// DMA_14903536517676
// MI455X (gfx1250) — matrix operand order repaired and checked by reading, not compiled, not run on hardware
//
#include <hip/hip_runtime.h>
#include <hip/hip_bf16.h>

// ---- problem constants (match reference) ----
constexpr int B_ = 2, S_ = 2048, D_ = 2048, H_ = 16, DH_ = 128;
#define FMIN_ (-3.4028234663852886e38f)
#define SCALE_ (0.08838834764831845f) // 1/sqrt(128)

typedef _Float16 f16;
typedef __attribute__((ext_vector_type(16))) _Float16 v16h;
typedef __attribute__((ext_vector_type(8)))  float    v8f;
typedef unsigned int u32x4 __attribute__((ext_vector_type(4)));
typedef unsigned int u32x8 __attribute__((ext_vector_type(8)));

union FragU { unsigned int u[8]; v16h h; };

// ---- WMMA fragment loaders (CDNA5 ISA 7.12.2 layouts, wave32) ----
// A-matrix 16x32 f16: lanes 0-15 row m=lane, k pairs {0..7,16..23}; lanes 16-31 same rows, k+8.
__device__ inline v16h load_a_frag(const f16* base, int ldk) {
  int l = threadIdx.x & 31;
  int m = l & 15, half = l >> 4;
  FragU f;
  const unsigned int* rp = (const unsigned int*)(base + m * ldk);
#pragma unroll
  for (int i = 0; i < 8; ++i) {
    int k = (i < 4 ? 2 * i : 16 + 2 * (i - 4)) + half * 8;
    f.u[i] = rp[k >> 1];
  }
  return f.h;
}
// B-matrix 32x16 f16 from a tile stored [n][k] row-major (i.e. W[N,K] / K[key][d]):
// lanes 0-15 col n=lane hold K {0..7,16..23}, lanes 16-31 hold K+8 (pairs per dword).
__device__ inline v16h load_b_frag(const f16* base, int ldk) {
  int l = threadIdx.x & 31;
  int n = l & 15, kh = l >> 4;
  FragU f;
  const unsigned int* rp = (const unsigned int*)(base + n * ldk);
#pragma unroll
  for (int i = 0; i < 8; ++i) f.u[i] = rp[((i < 4 ? 2 * i : 16 + 2 * (i - 4)) + kh * 8) >> 1];
  return f.h;
}

// ---- f32 -> f16 conversion ----
__global__ void f32_to_f16_kernel(const float* __restrict__ in, f16* __restrict__ out, long long n) {
  long long i = (long long)blockIdx.x * blockDim.x + threadIdx.x;
  long long st = (long long)gridDim.x * blockDim.x;
  for (; i < n; i += st) out[i] = (f16)in[i];
}

// ---- tiled WMMA GEMM: C[M,N] = A[M,K] @ W[N,K]^T + bias ----
// Block tile 64(M) x 256(N), 8 waves each owning 32x64 (2x4 WMMA tiles).
// MODE 0: f16 store scattered to [b,h,s,dh] (QKV). MODE 1: f32 row-major store.
template <int MODE>
__global__ void __launch_bounds__(256)
gemm_kernel(const f16* __restrict__ A, const f16* __restrict__ W,
            const float* __restrict__ bias, void* __restrict__ out,
            int M, int N, int K) {
  __shared__ __align__(16) f16 sA[64 * 32];
  __shared__ __align__(16) f16 sW[256 * 32];
  const int tid = threadIdx.x;
  const int wid = tid >> 5;
  const int wm = wid & 1, wn = wid >> 1;     // 2(M) x 4(N) waves, 32x64 each
  const int m0 = blockIdx.y * 64, n0 = blockIdx.x * 256;
  v8f acc[2][4] = {};
  for (int k0 = 0; k0 < K; k0 += 32) {
    __syncthreads();
    { // A tile 64x32: 256 x uint4
      int r = tid >> 2, c = (tid & 3) * 8;
      *(uint4*)(sA + r * 32 + c) = *(const uint4*)(A + (size_t)(m0 + r) * K + k0 + c);
    }
#pragma unroll
    for (int t = 0; t < 4; ++t) { // W tile 256x32: 1024 x uint4
      int idx = tid + t * 256;
      int r = idx >> 2, c = (idx & 3) * 8;
      *(uint4*)(sW + r * 32 + c) = *(const uint4*)(W + (size_t)(n0 + r) * K + k0 + c);
    }
    __syncthreads();
#pragma unroll
    for (int mi = 0; mi < 2; ++mi) {
      v16h af = load_a_frag(sA + (wm * 32 + mi * 16) * 32, 32);
#pragma unroll
      for (int ni = 0; ni < 4; ++ni) {
        v16h bf = load_b_frag(sW + (wn * 64 + ni * 16) * 32, 32);
        acc[mi][ni] = __builtin_amdgcn_wmma_f32_16x16x32_f16(
            false, af, false, bf, (short)0, acc[mi][ni], false, false);
      }
    }
  }
  // C/D layout: lanes 0-15 rows 0-7, lanes 16-31 rows 8-15; col = lane&15
  const int l = tid & 31;
  const int cn = l & 15, mb = (l >> 4) * 8;
#pragma unroll
  for (int mi = 0; mi < 2; ++mi)
#pragma unroll
    for (int ni = 0; ni < 4; ++ni)
#pragma unroll
      for (int i = 0; i < 8; ++i) {
        int gm = m0 + wm * 32 + mi * 16 + mb + i;
        int gn = n0 + wn * 64 + ni * 16 + cn;
        float v = acc[mi][ni][i] + bias[gn];
        if (MODE == 0) {
          int b = gm >> 11, s = gm & (S_ - 1);
          int h = gn >> 7, d = gn & (DH_ - 1);
          ((f16*)out)[(((size_t)b * H_ + h) * S_ + s) * DH_ + d] = (f16)v;
        } else {
          ((float*)out)[(size_t)gm * N + gn] = v;
        }
      }
}

// ---- dt = v_lin @ Wdt^T + bdt ; dmask = exp(A*softplus(dt)) -> [b,h,s] ----
__global__ void __launch_bounds__(256)
dt_mask_kernel(const f16* __restrict__ Vh, const f16* __restrict__ Wdt,
               const float* __restrict__ bdt, const float* __restrict__ Av,
               float* __restrict__ dmask) {
  const int lane = threadIdx.x & 31;
  long long task = (long long)blockIdx.x * 8 + (threadIdx.x >> 5);
  if (task >= (long long)B_ * S_ * H_) return;
  int h = (int)(task % H_);
  long long bs = task / H_;
  int s = (int)(bs % S_);
  int b = (int)(bs / S_);
  float sum = 0.f;
  for (int e = lane; e < D_; e += 32) {
    int hh = e >> 7, d = e & (DH_ - 1);
    sum += (float)Vh[(((size_t)b * H_ + hh) * S_ + s) * DH_ + d] *
           (float)Wdt[(size_t)h * D_ + e];
  }
#pragma unroll
  for (int m = 16; m >= 1; m >>= 1) sum += __shfl_xor(sum, m, 32);
  if (lane == 0) {
    float dt = sum + bdt[h];
    float sp = (dt > 20.f) ? dt : log1pf(__expf(dt));
    dmask[((size_t)b * H_ + h) * S_ + s] = __expf(Av[h] * sp);
  }
}

// ---- per-(b,h): bitonic sort, k-th value, combined column bias ----
__global__ void __launch_bounds__(1024)
topk_kernel(const float* __restrict__ dmask, const int* __restrict__ ratio,
            float* __restrict__ cbias) {
  __shared__ float s[S_];
  __shared__ float kth_s;
  const int row = blockIdx.x, tid = threadIdx.x, nt = blockDim.x;
  const float* src = dmask + (size_t)row * S_;
  for (int i = tid; i < S_; i += nt) s[i] = src[i];
  __syncthreads();
  const int rp = ratio[0];
  const int num = (int)((long long)S_ * rp / 1000);
  const bool do_topk = (rp > 0 && rp < 1000 && num > 0);
  if (do_topk) {
    for (unsigned k = 2; k <= (unsigned)S_; k <<= 1)
      for (unsigned j = k >> 1; j > 0; j >>= 1) {
        for (unsigned idx = tid; idx < (unsigned)S_; idx += nt) {
          unsigned ixj = idx ^ j;
          if (ixj > idx && ixj < (unsigned)S_) {
            float a = s[idx], bb = s[ixj];
            bool asc = ((idx & k) == 0);
            if ((a > bb) == asc) { s[idx] = bb; s[ixj] = a; }
          }
        }
        __syncthreads();
      }
    if (tid == 0) kth_s = s[num - 1];
    __syncthreads();
  }
  const float kth = do_topk ? kth_s : 0.f;
  for (int i = tid; i < S_; i += nt) {
    float v = src[i];
    cbias[(size_t)row * S_ + i] = (do_topk && v < kth) ? FMIN_ : v;
  }
}

// ---- flash attention: 4 waves/block share K/V tiles; 16 q-rows per wave ----
// K tile staged by the Tensor Data Mover (tensor_load_to_lds, TENSORcnt fence);
// V tile transposed in registers and stored as packed b32.
__global__ void __launch_bounds__(128)
attn_kernel(const f16* __restrict__ Qh, const f16* __restrict__ Kh,
            const f16* __restrict__ Vh, const float* __restrict__ cbias,
            f16* __restrict__ Oh) {
  __shared__ __align__(16) f16 sK[32 * 128];   // [key][d], filled by TDM
  __shared__ __align__(16) f16 sV[128 * 36];   // [d][key], padded stride
  __shared__ __align__(16) f16 sP[4][16 * 32]; // per-wave P staging
  const int tid = threadIdx.x;
  const int wid = tid >> 5, lane = tid & 31;
  const int bh = blockIdx.y;
  const int b = bh >> 4, h = bh & (H_ - 1);
  const int q0 = blockIdx.x * 64 + wid * 16;
  const int m = lane & 15, half = lane >> 4;
  const int gq = q0 + m;
  const size_t headoff = ((size_t)b * H_ + h) * S_;
  const unsigned ldsK = (unsigned)(unsigned long long)(const void*)sK; // low 32 bits = LDS offset

  // Q row fragments (4 K-chunks of 32), kept in registers for the whole loop
  v16h qf[4];
  {
    const unsigned int* qrow = (const unsigned int*)(Qh + (headoff + gq) * DH_);
#pragma unroll
    for (int c = 0; c < 4; ++c) {
      FragU f;
#pragma unroll
      for (int i = 0; i < 8; ++i) {
        int k = c * 32 + (i < 4 ? 2 * i : 16 + 2 * (i - 4)) + half * 8;
        f.u[i] = qrow[k >> 1];
      }
      qf[c] = f.h;
    }
  }

  v8f oacc[8] = {};
  float rmax[8], rsum[8];
#pragma unroll
  for (int i = 0; i < 8; ++i) { rmax[i] = -INFINITY; rsum[i] = 0.f; }

  const int ntiles = blockIdx.x * 2 + 2; // causal bound: keys <= q0_block+63
  for (int kt = 0; kt < ntiles; ++kt) {
    const int k0 = kt * 32;
    __syncthreads();
    if (wid == 0) {
      // TDM descriptor: 2D tile 128(d) x 32(keys), f16, row stride DH_, into sK.
      unsigned long long ga = (unsigned long long)(const void*)(Kh + (headoff + k0) * DH_);
      u32x4 g0 = { 1u,                                   // count=1 (valid descriptor)
                   ldsK,                                 // lds_addr
                   (unsigned)ga,                         // global_addr[31:0]
                   (unsigned)((ga >> 32) & 0x01FFFFFFull) | (2u << 30) }; // addr[56:32] | type=2
      u32x8 g1 = { 0x00010000u,                          // wg_mask=0, data_size=1 (2 bytes)
                   (unsigned)DH_ << 16,                  // tensor_dim0 = 128
                   (unsigned)S_ << 16,                   // tensor_dim1 = 2048
                   (unsigned)DH_ << 16,                  // tile_dim0 = 128
                   32u,                                  // tile_dim1 = 32
                   (unsigned)DH_,                        // tensor_dim0_stride = 128
                   0u, 0u };
      asm volatile("tensor_load_to_lds %0, %1" :: "s"(g0), "s"(g1) : "memory");
    }
    { // V tile transposed: pack key-pairs into dwords, 16 x b32 LDS stores/thread
      int rp2 = tid >> 3;                 // key pair index 0..15
      int c0 = (tid & 7) * 16;            // 16 d-columns
      const f16* v0 = Vh + (headoff + k0 + 2 * rp2) * DH_ + c0;
      union { uint4 q[2]; unsigned short us[16]; } ua, ub;
      ua.q[0] = *(const uint4*)(v0);
      ua.q[1] = *(const uint4*)(v0 + 8);
      ub.q[0] = *(const uint4*)(v0 + DH_);
      ub.q[1] = *(const uint4*)(v0 + DH_ + 8);
#pragma unroll
      for (int c = 0; c < 16; ++c)
        *(unsigned int*)(sV + (c0 + c) * 36 + 2 * rp2) =
            (unsigned)ua.us[c] | ((unsigned)ub.us[c] << 16);
    }
    if (wid == 0) __builtin_amdgcn_s_wait_tensorcnt(0);
    __syncthreads();

    // scores: two 16x16 tiles (keys k0..k0+15, k0+16..k0+31)
    float sc[2][8];
#pragma unroll
    for (int ni = 0; ni < 2; ++ni) {
      v8f a = {};
#pragma unroll
      for (int c = 0; c < 4; ++c) {
        v16h bf = load_b_frag(sK + (ni * 16) * 128 + c * 32, 128);
        a = __builtin_amdgcn_wmma_f32_16x16x32_f16(false, qf[c], false, bf,
                                                   (short)0, a, false, false);
      }
      const int gk = k0 + ni * 16 + m;
      const float cb = cbias[(size_t)bh * S_ + gk];
#pragma unroll
      for (int i = 0; i < 8; ++i) {
        int row = q0 + half * 8 + i;
        sc[ni][i] = a[i] * SCALE_ + ((gk > row) ? FMIN_ : cb);
      }
    }

    // online softmax update + stage P into LDS as f16
#pragma unroll
    for (int i = 0; i < 8; ++i) {
      float mx = fmaxf(sc[0][i], sc[1][i]);
#pragma unroll
      for (int xm = 8; xm >= 1; xm >>= 1) mx = fmaxf(mx, __shfl_xor(mx, xm, 32));
      float mnew = fmaxf(rmax[i], mx);
      float alpha = __expf(rmax[i] - mnew);
      float p0 = __expf(sc[0][i] - mnew);
      float p1 = __expf(sc[1][i] - mnew);
      float ps = p0 + p1;
#pragma unroll
      for (int xm = 8; xm >= 1; xm >>= 1) ps += __shfl_xor(ps, xm, 32);
      rsum[i] = rsum[i] * alpha + ps;
      rmax[i] = mnew;
#pragma unroll
      for (int t = 0; t < 8; ++t) oacc[t][i] *= alpha;
      sP[wid][(half * 8 + i) * 32 + m]      = (f16)p0;
      sP[wid][(half * 8 + i) * 32 + 16 + m] = (f16)p1;
    }
    __syncthreads();

    // O += P(16x32) @ V(32x128)
    v16h pa = load_a_frag(sP[wid], 32);
#pragma unroll
    for (int t = 0; t < 8; ++t) {
      v16h vb = load_b_frag(sV + (t * 16) * 36, 36);
      oacc[t] = __builtin_amdgcn_wmma_f32_16x16x32_f16(false, pa, false, vb,
                                                       (short)0, oacc[t], false, false);
    }
  }

  // normalize and store O as [b, s, h, d] (== [b, s, D] row-major for Wo GEMM)
#pragma unroll
  for (int i = 0; i < 8; ++i) {
    float inv = (rsum[i] > 0.f) ? 1.f / rsum[i] : 0.f;
    int row = q0 + half * 8 + i;
#pragma unroll
    for (int t = 0; t < 8; ++t) {
      int d = t * 16 + m;
      Oh[((size_t)b * S_ + row) * D_ + h * DH_ + d] = (f16)(oacc[t][i] * inv);
    }
  }
}

extern "C" void kernel_launch(void* const* d_in, const int* in_sizes, int n_in,
                              void* d_out, int out_size, void* d_ws, size_t ws_size,
                              hipStream_t stream) {
  (void)in_sizes; (void)n_in; (void)out_size; (void)ws_size;
  const float* hidden = (const float*)d_in[0];
  // d_in[1] attention_mask is the deterministic causal mask; applied analytically.
  const float* Wq = (const float*)d_in[2];  const float* bq  = (const float*)d_in[3];
  const float* Wk = (const float*)d_in[4];  const float* bk  = (const float*)d_in[5];
  const float* Wv = (const float*)d_in[6];  const float* bv  = (const float*)d_in[7];
  const float* Wdt = (const float*)d_in[8]; const float* bdt = (const float*)d_in[9];
  const float* Av = (const float*)d_in[10];
  const float* Wo = (const float*)d_in[11]; const float* bo  = (const float*)d_in[12];
  const int* ratio = (const int*)d_in[13];
  float* out = (float*)d_out;

  size_t off = 0;
  char* ws = (char*)d_ws;
  auto alloc = [&](size_t nbytes) -> void* {
    void* p = ws + off;
    off += (nbytes + 255) & ~(size_t)255;
    return p;
  };
  f16* hid_h = (f16*)alloc((size_t)B_ * S_ * D_ * 2);
  f16* wq_h  = (f16*)alloc((size_t)D_ * D_ * 2);
  f16* wk_h  = (f16*)alloc((size_t)D_ * D_ * 2);
  f16* wv_h  = (f16*)alloc((size_t)D_ * D_ * 2);
  f16* wo_h  = (f16*)alloc((size_t)D_ * D_ * 2);
  f16* wdt_h = (f16*)alloc((size_t)H_ * D_ * 2);
  f16* q_h   = (f16*)alloc((size_t)B_ * S_ * D_ * 2);
  f16* k_h   = (f16*)alloc((size_t)B_ * S_ * D_ * 2);
  f16* v_h   = (f16*)alloc((size_t)B_ * S_ * D_ * 2);
  f16* o_h   = (f16*)alloc((size_t)B_ * S_ * D_ * 2);
  float* dmask = (float*)alloc((size_t)B_ * H_ * S_ * 4);
  float* cbias = (float*)alloc((size_t)B_ * H_ * S_ * 4);

  // 1) precision conversion (done once; all GEMMs consume f16)
  f32_to_f16_kernel<<<2048, 256, 0, stream>>>(hidden, hid_h, (long long)B_ * S_ * D_);
  f32_to_f16_kernel<<<2048, 256, 0, stream>>>(Wq, wq_h, (long long)D_ * D_);
  f32_to_f16_kernel<<<2048, 256, 0, stream>>>(Wk, wk_h, (long long)D_ * D_);
  f32_to_f16_kernel<<<2048, 256, 0, stream>>>(Wv, wv_h, (long long)D_ * D_);
  f32_to_f16_kernel<<<2048, 256, 0, stream>>>(Wo, wo_h, (long long)D_ * D_);
  f32_to_f16_kernel<<<128, 256, 0, stream>>>(Wdt, wdt_h, (long long)H_ * D_);

  // 2) QKV projections (WMMA), head-scattered f16 outputs
  dim3 gg(D_ / 256, (B_ * S_) / 64);
  gemm_kernel<0><<<gg, 256, 0, stream>>>(hid_h, wq_h, bq, q_h, B_ * S_, D_, D_);
  gemm_kernel<0><<<gg, 256, 0, stream>>>(hid_h, wk_h, bk, k_h, B_ * S_, D_, D_);
  gemm_kernel<0><<<gg, 256, 0, stream>>>(hid_h, wv_h, bv, v_h, B_ * S_, D_, D_);

  // 3) dynamic mask, 4) top-k threshold -> combined column bias
  dt_mask_kernel<<<8192, 256, 0, stream>>>(v_h, wdt_h, bdt, Av, dmask);
  topk_kernel<<<B_ * H_, 1024, 0, stream>>>(dmask, ratio, cbias);

  // 5) flash attention (WMMA QK^T and PV, TDM-staged K tiles)
  attn_kernel<<<dim3(S_ / 64, B_ * H_), 128, 0, stream>>>(q_h, k_h, v_h, cbias, o_h);

  // 6) output projection (WMMA), f32 store to d_out
  gemm_kernel<1><<<gg, 256, 0, stream>>>(o_h, wo_h, bo, out, B_ * S_, D_, D_);
}
